// RayNetwork_62629213110628
// MI455X (gfx1250) — matrix operand order repaired and checked by reading, not compiled, not run on hardware
//
#include <hip/hip_runtime.h>

// ---------------------------------------------------------------------------
// Problem constants (match reference)
// ---------------------------------------------------------------------------
#define NPTS   32768
#define FEAT   384
#define CTX    3840
#define EPSV   1e-5f

// WMMA tiling
#define KT  (CTX / 32)    // 120 k-tiles of 32
#define NT  (FEAT / 16)   // 24 n-tiles of 16
#define MB  64            // points (M rows) per workgroup
#define AST 136           // LDS A-panel row stride in bf16 (128 + 8 pad -> bank-conflict-free)

// workspace layout (bytes)
#define B_BLK_BYTES 1024                              // one 32x16 bf16 B block
#define B_BYTES     ((size_t)KT * NT * B_BLK_BYTES)   // 2,949,120
#define PF_OFF      B_BYTES
#define PF_BYTES    ((size_t)NPTS * FEAT * 4)         // 48 MiB
#define ST_OFF      (PF_OFF + PF_BYTES)
// stats: [0..31]=sum  [32..63]=sumsq  [64..95]=mu  [96..127]=rsqrt

typedef __attribute__((ext_vector_type(16))) __bf16          v16bf;
typedef __attribute__((ext_vector_type(8)))  float           v8f;
typedef __attribute__((ext_vector_type(8)))  unsigned short  v8u;
typedef __attribute__((ext_vector_type(16))) unsigned short  v16u;

__device__ __forceinline__ unsigned short f2bf(float f) {
    union { float f; unsigned u; } c; c.f = f;
    unsigned u = c.u;
    return (unsigned short)((u + 0x7FFFu + ((u >> 16) & 1u)) >> 16); // RNE
}

// ---------------------------------------------------------------------------
// Kernel 0: repack W_red (f32 [384][3840]) into bf16 WMMA B-fragment blocks.
// Block (kt, nt): 32 lanes x 16 bf16; lane n-col = lane&15,
// lane k-range = kt*32 + (lane>>4)*16 + j  (VGPR v packs K=2v,2v+1).
// ---------------------------------------------------------------------------
__global__ void prep_B(const float* __restrict__ Wred,
                       unsigned short* __restrict__ B) {
    int gid  = blockIdx.x * blockDim.x + threadIdx.x;   // 92160 threads
    int blk  = gid >> 5;
    int lane = gid & 31;
    int kt = blk / NT, nt = blk % NT;
    int n     = nt * 16 + (lane & 15);
    int kbase = kt * 32 + ((lane >> 4) << 4);
    unsigned short* dst = B + (size_t)gid * 16;
#pragma unroll
    for (int j = 0; j < 16; ++j)
        dst[j] = f2bf(Wred[(size_t)n * CTX + kbase + j]);
}

// ---------------------------------------------------------------------------
// Kernel 1: zero groupnorm accumulators (must run every call: graph replay)
// ---------------------------------------------------------------------------
__global__ void zero_stats(float* __restrict__ stats) {
    if (threadIdx.x < 64) stats[threadIdx.x] = 0.0f;
}

// ---------------------------------------------------------------------------
// Kernel 2: fused project + bilinear-gather + bf16 WMMA reduce-GEMM
//           + xyz linear + bias + partial groupnorm stats.
// One block = 64 points (4 m-tiles), 256 threads = 8 waves.
// Wave w owns n-tiles {3w, 3w+1, 3w+2} x all 4 m-tiles = 12 accumulators.
// B fragments are loaded once per k-tile and reused across 4 m-tiles,
// cutting W_red L2 traffic 4x vs a 16-row block.
// ---------------------------------------------------------------------------
__global__ void __launch_bounds__(256) fused_main(
    const float* __restrict__ x,     const float* __restrict__ Kmat,
    const float* __restrict__ Wxyz,  const float* __restrict__ bxyz,
    const float* __restrict__ bred,
    const float* __restrict__ f0, const float* __restrict__ f1,
    const float* __restrict__ f2, const float* __restrict__ f3,
    const unsigned short* __restrict__ B,
    float* __restrict__ pf, float* __restrict__ stats)
{
    __shared__ float          s_x[MB][3];
    __shared__ int            s_off[4][MB][4];   // float4-granule corner offsets
    __shared__ float          s_w[4][MB][4];     // bilinear weights
    __shared__ unsigned short s_A[MB * AST];     // bf16 A panel, padded rows
    __shared__ float          s_gs[32], s_gq[32];

    const int tid = threadIdx.x;
    const int m0  = blockIdx.x * MB;

    if (tid < 32) { s_gs[tid] = 0.0f; s_gq[tid] = 0.0f; }

    if (tid < MB) {
        float xx = x[(m0 + tid) * 3 + 0];
        float xy = x[(m0 + tid) * 3 + 1];
        float xz = x[(m0 + tid) * 3 + 2];
        s_x[tid][0] = xx; s_x[tid][1] = xy; s_x[tid][2] = xz;
        // pinhole projection p = K @ x ; uv = p[:2] / (p[2]+1e-3)
        float p0 = xx * Kmat[0] + xy * Kmat[1] + xz * Kmat[2];
        float p1 = xx * Kmat[3] + xy * Kmat[4] + xz * Kmat[5];
        float p2 = xx * Kmat[6] + xy * Kmat[7] + xz * Kmat[8];
        float inv = 1.0f / (p2 + 1e-3f);
        float h01 = p0 * inv;                 // row coord (x H)
        float w01 = p1 * inv;                 // col coord (x W)
        const int HH[4] = {64, 32, 16, 8};
        const int WW[4] = {64, 32, 16, 8};
        const int CC[4] = {256, 512, 1024, 2048};
#pragma unroll
        for (int l = 0; l < 4; ++l) {
            float yc = h01 * (float)HH[l];
            float xc = w01 * (float)WW[l];
            float yf = floorf(yc), xf = floorf(xc);
            float fy = yc - yf,   fx = xc - xf;
            int y0 = (int)yf,     xq = (int)xf;
            int y0c = min(max(y0, 0),     HH[l] - 1);
            int y1c = min(max(y0 + 1, 0), HH[l] - 1);
            int x0c = min(max(xq, 0),     WW[l] - 1);
            int x1c = min(max(xq + 1, 0), WW[l] - 1);
            // offsets in float4 units (C is a multiple of 4)
            s_off[l][tid][0] = ((y0c * WW[l] + x0c) * CC[l]) >> 2;
            s_off[l][tid][1] = ((y0c * WW[l] + x1c) * CC[l]) >> 2;
            s_off[l][tid][2] = ((y1c * WW[l] + x0c) * CC[l]) >> 2;
            s_off[l][tid][3] = ((y1c * WW[l] + x1c) * CC[l]) >> 2;
            s_w[l][tid][0] = (1.0f - fy) * (1.0f - fx);
            s_w[l][tid][1] = (1.0f - fy) * fx;
            s_w[l][tid][2] = fy * (1.0f - fx);
            s_w[l][tid][3] = fy * fx;
        }
    }
    __syncthreads();

    const float* fl[4] = {f0, f1, f2, f3};

    v8f acc[3][4] = {};
    const int wv   = tid >> 5;
    const int lane = tid & 31;
    const int hi   = lane >> 4;
    const int mrow = lane & 15;

    for (int ck = 0; ck < 30; ++ck) {
        const int kb  = ck * 128;                       // chunk never straddles levels
        const int lvl = (kb < 256) ? 0 : (kb < 768) ? 1 : (kb < 1792) ? 2 : 3;
        const int lst = (lvl == 0) ? 0 : (lvl == 1) ? 256 : (lvl == 2) ? 768 : 1792;
        const float* fp  = fl[lvl];
        const int cbase  = kb - lst;

        // ---- cooperative bilinear gather of MBx128 panel -> bf16 LDS ----
#pragma unroll
        for (int i = 0; i < (MB * 32) / 256; ++i) {     // 8 float4-quads / thread
            int q  = tid + i * 256;
            int m  = q >> 5;                            // 32 quads per row
            int cc = (q & 31) * 4;
            const float4* bp = (const float4*)(fp + cbase + cc);
            float4 t00 = bp[s_off[lvl][m][0]];
            float4 t01 = bp[s_off[lvl][m][1]];
            float4 t10 = bp[s_off[lvl][m][2]];
            float4 t11 = bp[s_off[lvl][m][3]];
            float w00 = s_w[lvl][m][0], w01 = s_w[lvl][m][1];
            float w10 = s_w[lvl][m][2], w11 = s_w[lvl][m][3];
            unsigned short* dst = &s_A[m * AST + cc];
            dst[0] = f2bf(w00 * t00.x + w01 * t01.x + w10 * t10.x + w11 * t11.x);
            dst[1] = f2bf(w00 * t00.y + w01 * t01.y + w10 * t10.y + w11 * t11.y);
            dst[2] = f2bf(w00 * t00.z + w01 * t01.z + w10 * t10.z + w11 * t11.z);
            dst[3] = f2bf(w00 * t00.w + w01 * t01.w + w10 * t10.w + w11 * t11.w);
        }
        __syncthreads();

        // ---- 4 k-subtiles of 32: 3 B frags, reused across 4 m-tiles ----
#pragma unroll
        for (int k4 = 0; k4 < 4; ++k4) {
            const int kl = k4 * 32;
            const int kt = ck * 4 + k4;
            const unsigned short* bb =
                B + ((size_t)(kt * NT + wv * 3) * 32 + lane) * 16;
            __builtin_prefetch(bb + (size_t)NT * 32 * 16, 0, 1);  // next k-tile

            v16bf b0 = __builtin_bit_cast(v16bf, *(const v16u*)(bb));
            v16bf b1 = __builtin_bit_cast(v16bf, *(const v16u*)(bb + 512));
            v16bf b2 = __builtin_bit_cast(v16bf, *(const v16u*)(bb + 1024));

#pragma unroll
            for (int mt = 0; mt < 4; ++mt) {
                // A fragment in contiguous k order, same as the prep_B blocks:
                // lane<16 holds K {kl..kl+15}; lane>=16 holds K {kl+16..kl+31}.
                const int arow = (mt * 16 + mrow) * AST;
                v8u alo = *(const v8u*)&s_A[arow + kl + hi * 16];
                v8u ahi = *(const v8u*)&s_A[arow + kl + hi * 16 + 8];
                v16u au = __builtin_shufflevector(alo, ahi,
                             0, 1, 2, 3, 4, 5, 6, 7, 8, 9, 10, 11, 12, 13, 14, 15);
                v16bf a = __builtin_bit_cast(v16bf, au);

                acc[0][mt] = __builtin_amdgcn_wmma_f32_16x16x32_bf16(
                                 false, a, false, b0, (short)0, acc[0][mt], false, false);
                acc[1][mt] = __builtin_amdgcn_wmma_f32_16x16x32_bf16(
                                 false, a, false, b1, (short)0, acc[1][mt], false, false);
                acc[2][mt] = __builtin_amdgcn_wmma_f32_16x16x32_bf16(
                                 false, a, false, b2, (short)0, acc[2][mt], false, false);
            }
        }
        __syncthreads();
    }

    // ---- epilogue: + xyz linear + biases, store pf, groupnorm partials ----
#pragma unroll
    for (int t = 0; t < 3; ++t) {
        const int n = (wv * 3 + t) * 16 + mrow;       // output channel
        const int g = n / 12;                         // groupnorm group
        float w0 = Wxyz[n * 3 + 0], w1 = Wxyz[n * 3 + 1], w2 = Wxyz[n * 3 + 2];
        float bb = bred[n] + bxyz[n];
        float lsum = 0.0f, lsq = 0.0f;
#pragma unroll
        for (int mt = 0; mt < 4; ++mt) {
            v8f a = acc[t][mt];
#pragma unroll
            for (int r = 0; r < 8; ++r) {
                const int m = mt * 16 + r + hi * 8;   // C-layout row
                float val = a[r] + bb
                          + s_x[m][0] * w0 + s_x[m][1] * w1 + s_x[m][2] * w2;
                pf[(size_t)(m0 + m) * FEAT + n] = val;
                lsum += val; lsq += val * val;
            }
        }
        atomicAdd(&s_gs[g], lsum);
        atomicAdd(&s_gq[g], lsq);
    }
    __syncthreads();
    if (tid < 32) {
        atomicAdd(&stats[tid],      s_gs[tid]);
        atomicAdd(&stats[32 + tid], s_gq[tid]);
    }
}

// ---------------------------------------------------------------------------
// Kernel 3: finalize groupnorm statistics
// ---------------------------------------------------------------------------
__global__ void finalize_stats(float* __restrict__ stats) {
    int g = threadIdx.x;
    if (g < 32) {
        const float cnt = 12.0f * (float)NPTS;
        float mu  = stats[g] / cnt;
        float var = stats[32 + g] / cnt - mu * mu;
        stats[64 + g] = mu;
        stats[96 + g] = rsqrtf(var + EPSV);
    }
}

// ---------------------------------------------------------------------------
// Kernel 4: normalize + 384->3 output projection
// ---------------------------------------------------------------------------
__global__ void __launch_bounds__(256) out_proj(
    const float* __restrict__ pf, const float* __restrict__ stats,
    const float* __restrict__ Wout, const float* __restrict__ bout,
    float* __restrict__ out)
{
    __shared__ float sW[3 * FEAT];
    __shared__ float smu[32], srs[32];
    const int tid = threadIdx.x;
    for (int i = tid; i < 3 * FEAT; i += 256) sW[i] = Wout[i];
    if (tid < 32) { smu[tid] = stats[64 + tid]; srs[tid] = stats[96 + tid]; }
    __syncthreads();

    const int m = blockIdx.x * 256 + tid;
    const float* row = pf + (size_t)m * FEAT;
    float a0 = 0.0f, a1 = 0.0f, a2 = 0.0f;
#pragma unroll 4
    for (int n = 0; n < FEAT; ++n) {
        const int g = n / 12;
        float xn = (row[n] - smu[g]) * srs[g];
        a0 += xn * sW[n];
        a1 += xn * sW[FEAT + n];
        a2 += xn * sW[2 * FEAT + n];
    }
    out[m * 3 + 0] = a0 + bout[0];
    out[m * 3 + 1] = a1 + bout[1];
    out[m * 3 + 2] = a2 + bout[2];
}

// ---------------------------------------------------------------------------
// Host launcher
// ---------------------------------------------------------------------------
extern "C" void kernel_launch(void* const* d_in, const int* in_sizes, int n_in,
                              void* d_out, int out_size, void* d_ws, size_t ws_size,
                              hipStream_t stream) {
    // setup_inputs dict order:
    // 0:t 1:x 2:K 3:W_xyz 4:b_xyz 5:W_red 6:b_red 7:W_out 8:b_out 9..12:f0..f3
    const float* x    = (const float*)d_in[1];
    const float* Kmat = (const float*)d_in[2];
    const float* Wxyz = (const float*)d_in[3];
    const float* bxyz = (const float*)d_in[4];
    const float* Wred = (const float*)d_in[5];
    const float* bred = (const float*)d_in[6];
    const float* Wout = (const float*)d_in[7];
    const float* bout = (const float*)d_in[8];
    const float* f0   = (const float*)d_in[9];
    const float* f1   = (const float*)d_in[10];
    const float* f2   = (const float*)d_in[11];
    const float* f3   = (const float*)d_in[12];

    unsigned short* Bw    = (unsigned short*)d_ws;
    float*          pf    = (float*)((char*)d_ws + PF_OFF);
    float*          stats = (float*)((char*)d_ws + ST_OFF);
    float*          out   = (float*)d_out;

    prep_B        <<<(KT * NT * 32) / 256, 256, 0, stream>>>(Wred, Bw);
    zero_stats    <<<1, 64, 0, stream>>>(stats);
    fused_main    <<<NPTS / MB, 256, 0, stream>>>(x, Kmat, Wxyz, bxyz, bred,
                                                  f0, f1, f2, f3, Bw, pf, stats);
    finalize_stats<<<1, 64, 0, stream>>>(stats);
    out_proj      <<<NPTS / 256, 256, 0, stream>>>(pf, stats, Wout, bout, out);
}
